// EfficientFormerV2Attention4D_14190571946621
// MI455X (gfx1250) — matrix operand order repaired and checked by reading, not compiled, not run on hardware
//
#include <hip/hip_runtime.h>
#include <hip/hip_bf16.h>
#include <math.h>

// ---------------- problem constants ----------------
#define BB   128
#define DIMC 256
#define RES  14
#define NPIX 196          // RES*RES
#define HH   8
#define KDIM 32
#define DVAL 128
#define DHC  1024         // HH*DVAL
#define N2   208          // NPIX padded to 13*16 (row/col tiles for logits)
#define NP   224          // NPIX padded to 7*32  (K-dim for attn@V, pitches)
#define SCALE 0.17677669529663689f  // 1/sqrt(32)

typedef __attribute__((ext_vector_type(16))) __bf16 v16bf;
typedef __attribute__((ext_vector_type(8)))  __bf16 v8bf;
typedef __attribute__((ext_vector_type(8)))  float  v8f;

static __device__ inline v8f wmma_bf16(v16bf a, v16bf b, v8f c) {
  // v_wmma_f32_16x16x32_bf16  D = A(16x32) * B(32x16) + C
  return __builtin_amdgcn_wmma_f32_16x16x32_bf16(false, a, false, b, (short)0, c,
                                                 false, false);
}

// A-fragment: lane holds K chunks [base..base+7] and [base+16..base+23]
// (base already applied by caller) -> two 16-byte vector loads.
static __device__ inline v16bf ldA(const __bf16* p) {
  v8bf lo = *(const v8bf*)p;
  v8bf hi = *(const v8bf*)(p + 16);
  return __builtin_shufflevector(lo, hi, 0, 1, 2, 3, 4, 5, 6, 7,
                                 8, 9, 10, 11, 12, 13, 14, 15);
}
// B-fragment: same K chunks as A ([base..base+7], [base+16..base+23]).
static __device__ inline v16bf ldB(const __bf16* p) {
  return ldA(p);
}

// ======================================================================
// Prep kernels: build fragment-native bf16 operand layouts.
// ======================================================================
__global__ __launch_bounds__(256) void prep_wp_kernel(const float* __restrict__ w_p,
                                                      __bf16* __restrict__ wpb) {
  int idx = blockIdx.x * 256 + threadIdx.x;         // idx = o*1024 + ch
  int o = idx >> 10, ch = idx & 1023;
  wpb[idx] = (__bf16)w_p[ch * DIMC + o];
}

__global__ __launch_bounds__(256) void prep_wqkv_kernel(
    const float* __restrict__ w_q, const float* __restrict__ w_k,
    const float* __restrict__ w_v, __bf16* __restrict__ wqkv) {
  int idx = blockIdx.x * 256 + threadIdx.x;         // idx = o*256 + c
  int o = idx >> 8, c = idx & 255;
  float v;
  if (o < 256)      v = w_q[c * 256 + o];
  else if (o < 512) v = w_k[c * 256 + (o - 256)];
  else              v = w_v[c * 1024 + (o - 512)];
  wqkv[idx] = (__bf16)v;
}

__global__ __launch_bounds__(256) void prep_x_kernel(const float* __restrict__ hidden,
                                                     __bf16* __restrict__ xb) {
  int idx = blockIdx.x * 256 + threadIdx.x;         // idx = ((b*208)+p)*256 + c
  int c = idx & 255;
  int p = (idx >> 8) % N2;
  int b = idx / (N2 * 256);
  xb[idx] = (p < NPIX) ? (__bf16)hidden[((size_t)b * DIMC + c) * NPIX + p]
                       : (__bf16)0.f;
}

__global__ __launch_bounds__(256) void prep_vpad_kernel(__bf16* __restrict__ vb) {
  int idx = blockIdx.x * 256 + threadIdx.x;         // zero pad cols [196,224)
  int col = NPIX + idx % (NP - NPIX);
  int row = idx / (NP - NPIX);                      // row = b*DHC + ch
  vb[(size_t)row * NP + col] = (__bf16)0.f;
}

// ======================================================================
// Kernel 1: fused QKV projection + cbn via WMMA.
// q/k -> bf16 [b][h][p(0..207)][kd]; v -> f32 vimg[b][ch][p] + bf16 vb[b][ch][224]
// grid (13 pixel tiles, 12 otile-groups, 128 batch), block 256 (8 waves)
// ======================================================================
__global__ __launch_bounds__(256) void qkv_kernel(
    const __bf16* __restrict__ xb, const __bf16* __restrict__ wqkv,
    const float* __restrict__ b_q, const float* __restrict__ s_q,
    const float* __restrict__ t_q, const float* __restrict__ b_k,
    const float* __restrict__ s_k, const float* __restrict__ t_k,
    const float* __restrict__ b_v, const float* __restrict__ s_v,
    const float* __restrict__ t_v, __bf16* __restrict__ qb,
    __bf16* __restrict__ kb, float* __restrict__ vimg,
    __bf16* __restrict__ vb) {
  const int lane = threadIdx.x & 31;
  const int wv   = threadIdx.x >> 5;
  const int l15  = lane & 15;
  const int hi16 = (lane >> 4) & 1;
  const int abase = hi16 * 8;

  const int pt = blockIdx.x;                 // pixel tile (16 pixels)
  const int ot = blockIdx.y * 8 + wv;        // output-channel tile (16 chans)
  const int b  = blockIdx.z;
  const int p  = pt * 16 + l15;              // B col / D col

  const __bf16* arow = wqkv + (size_t)(ot * 16 + l15) * DIMC;
  const __bf16* xrow = xb + ((size_t)b * N2 + p) * DIMC;

  v8f acc = {0.f, 0.f, 0.f, 0.f, 0.f, 0.f, 0.f, 0.f};
  for (int kt = 0; kt < DIMC / 32; ++kt) {
    v16bf af = ldA(arow + kt * 32 + abase);
    v16bf bf = ldB(xrow + kt * 32 + abase);
    acc = wmma_bf16(af, bf, acc);
  }

  const int og = ot * 16;                    // global o base of this tile
  if (ot < 32) {                             // ---- q or k ----
    const bool isq = (ot < 16);
    const float* bias = isq ? b_q : b_k;
    const float* sc   = isq ? s_q : s_k;
    const float* sh   = isq ? t_q : t_k;
    const int ob = isq ? og : og - 256;      // local o base (0..255)
    const int ol0 = ob + hi16 * 8;
    const int h = ol0 >> 5, kd0 = ol0 & 31;
    v8bf qv;
#pragma unroll
    for (int r = 0; r < 8; ++r) {
      int ol = ol0 + r;
      qv[r] = (__bf16)((acc[r] + bias[ol]) * sc[ol] + sh[ol]);
    }
    __bf16* dst = (isq ? qb : kb) + (((size_t)b * HH + h) * N2 + p) * KDIM + kd0;
    *(v8bf*)dst = qv;
  } else {                                   // ---- v ----
    const int ob = og - 512;                 // local ch base (0..1023)
#pragma unroll
    for (int r = 0; r < 8; ++r) {
      int ch = ob + r + hi16 * 8;
      float y = (acc[r] + b_v[ch]) * s_v[ch] + t_v[ch];
      if (p < NPIX) {
        vimg[((size_t)b * DHC + ch) * NPIX + p] = y;
        vb[((size_t)b * DHC + ch) * NP + p] = (__bf16)y;
      }
    }
  }
}

// ======================================================================
// Kernel 2: fused attention per (b, 16-query-row tile). 8 waves = 8 heads.
// LDS: attnA f32[8][16][224] | attnB f32[8][16][224] | attnC bf16[8][16][224]
//      xbuf bf16[16][1024] aliases attnA (free after th1).   total 280 KB
// ======================================================================
__global__ __launch_bounds__(256) void attn_kernel(
    const __bf16* __restrict__ qb, const __bf16* __restrict__ kb,
    const float* __restrict__ vimg, const __bf16* __restrict__ vb,
    const __bf16* __restrict__ wpb,
    const int* __restrict__ bias_idxs, const float* __restrict__ attn_biases,
    int noff,
    const float* __restrict__ w_th1, const float* __restrict__ b_th1,
    const float* __restrict__ w_th2, const float* __restrict__ b_th2,
    const float* __restrict__ w_vl, const float* __restrict__ b_vl,
    const float* __restrict__ s_vl, const float* __restrict__ t_vl,
    const float* __restrict__ b_p, const float* __restrict__ s_p,
    const float* __restrict__ t_p, float* __restrict__ out) {
  extern __shared__ char smem[];
  float*  attnA = (float*)smem;                       // 114688 B
  float*  attnB = (float*)(smem + 114688);            // 114688 B
  __bf16* attnC = (__bf16*)(smem + 229376);           // 57344 B
  __bf16* xbuf  = (__bf16*)smem;                      // aliases attnA (32 KB)

  const int tid  = threadIdx.x;
  const int lane = tid & 31;
  const int wv   = tid >> 5;                           // wave == head
  const int l15  = lane & 15;
  const int hi16 = (lane >> 4) & 1;
  const int nt   = blockIdx.x;                         // query-row tile
  const int b    = blockIdx.y;
  const int h    = wv;
  const int abase = hi16 * 8;

  // ---------- phase A: logits = q @ k * SCALE + gathered bias ----------
  {
    const __bf16* qrow = qb + (((size_t)b * HH + h) * N2 + nt * 16 + l15) * KDIM;
    v16bf aq = ldA(qrow + abase);
    for (int mt = 0; mt < 13; ++mt) {
      const int mcol = mt * 16 + l15;
      const __bf16* krow = kb + (((size_t)b * HH + h) * N2 + mcol) * KDIM;
      v16bf bk = ldB(krow + abase);
      v8f acc = {0.f, 0.f, 0.f, 0.f, 0.f, 0.f, 0.f, 0.f};
      acc = wmma_bf16(aq, bk, acc);
#pragma unroll
      for (int r = 0; r < 8; ++r) {
        int nl = r + hi16 * 8;
        int n  = nt * 16 + nl;
        float v = 0.f;
        if (n < NPIX && mcol < NPIX) {
          int idx = bias_idxs[n * NPIX + mcol];
          v = acc[r] * SCALE + attn_biases[h * noff + idx];
        }
        attnA[(h * 16 + nl) * NP + mcol] = v;
      }
    }
  }
  __syncthreads();

  // ---------- talking heads 1 (mix over h), mask pad cols to -inf ----------
  for (int i = tid; i < HH * 16 * NP; i += 256) {
    int mm = i % NP, nl = (i / NP) & 15, g = i / (NP * 16);
    float v;
    if (mm < NPIX) {
      float s = b_th1[g];
#pragma unroll
      for (int h2 = 0; h2 < HH; ++h2)
        s += w_th1[g * HH + h2] * attnA[(h2 * 16 + nl) * NP + mm];
      v = s;
    } else {
      v = -3.0e38f;
    }
    attnB[(g * 16 + nl) * NP + mm] = v;
  }
  __syncthreads();

  // ---------- softmax over m (one thread per (g,n) row) ----------
  if (tid < 128) {
    float* row = attnB + tid * NP;           // tid = g*16+nl
    float mx = -3.0e38f;
    for (int m = 0; m < NPIX; ++m) mx = fmaxf(mx, row[m]);
    float s = 0.f;
    for (int m = 0; m < NPIX; ++m) { float e = __expf(row[m] - mx); row[m] = e; s += e; }
    float inv = 1.f / s;
    for (int m = 0; m < NPIX; ++m) row[m] *= inv;
  }
  __syncthreads();

  // ---------- talking heads 2 -> bf16 attnC (A-fragment layout) ----------
  for (int i = tid; i < HH * 16 * NP; i += 256) {
    int mm = i % NP, nl = (i / NP) & 15, g = i / (NP * 16);
    float v = 0.f;
    if (mm < NPIX) {
      float s = b_th2[g];
#pragma unroll
      for (int h2 = 0; h2 < HH; ++h2)
        s += w_th2[g * HH + h2] * attnB[(h2 * 16 + nl) * NP + mm];
      v = s;
    }
    attnC[(g * 16 + nl) * NP + mm] = (__bf16)v;
  }
  __syncthreads();

  // ---------- attn @ V, + depthwise-3x3 vl + GELU -> xbuf(bf16) ----------
  for (int dvt = 0; dvt < DVAL / 16; ++dvt) {
    v8f acc = {0.f, 0.f, 0.f, 0.f, 0.f, 0.f, 0.f, 0.f};
    const int chB = h * DVAL + dvt * 16 + l15;
    const __bf16* vrowB = vb + ((size_t)b * DHC + chB) * NP;
    const __bf16* arow  = attnC + (h * 16 + l15) * NP;
    for (int kt = 0; kt < NP / 32; ++kt) {
      v16bf af = ldA(arow + kt * 32 + abase);         // ds_load_b128 x2
      v16bf bf = ldB(vrowB + kt * 32 + abase);        // global_load_b128 x2
      acc = wmma_bf16(af, bf, acc);
    }
#pragma unroll
    for (int r = 0; r < 8; ++r) {
      int nl = r + hi16 * 8;
      int p  = nt * 16 + nl;
      int ch = h * DVAL + dvt * 16 + l15;
      float val = acc[r];
      if (p < NPIX) {
        // depthwise 3x3 conv (vl) + its cbn, computed inline
        int py = p / RES, px = p % RES;
        const float* vrow = vimg + ((size_t)b * DHC + ch) * NPIX;
        const float* wl = w_vl + ch * 9;
        float s = 0.f;
#pragma unroll
        for (int ky = 0; ky < 3; ++ky) {
          int yy = py + ky - 1;
          if (yy < 0 || yy >= RES) continue;
#pragma unroll
          for (int kx = 0; kx < 3; ++kx) {
            int xx = px + kx - 1;
            if (xx < 0 || xx >= RES) continue;
            s += vrow[yy * RES + xx] * wl[ky * 3 + kx];
          }
        }
        val += (s + b_vl[ch]) * s_vl[ch] + t_vl[ch];
      }
      // exact GELU: 0.5*x*(1+erf(x/sqrt(2)))
      float g = 0.5f * val * (1.0f + erff(val * 0.70710678118654752f));
      xbuf[nl * DHC + ch] = (__bf16)g;
    }
  }
  __syncthreads();

  // ---------- output projection: (16 x 1024) @ (1024 x 256) + cbn ----------
  for (int q = 0; q < 2; ++q) {
    const int ot = wv * 2 + q;               // 16 output tiles over 8 waves
    const int o  = ot * 16 + l15;
    const __bf16* wr = wpb + (size_t)o * DHC;
    const __bf16* xr = xbuf + l15 * DHC;
    v8f acc = {0.f, 0.f, 0.f, 0.f, 0.f, 0.f, 0.f, 0.f};
    for (int kt = 0; kt < DHC / 32; ++kt) {
      if (kt + 2 < DHC / 32)
        __builtin_prefetch(wr + (kt + 2) * 32, 0, 0);  // global_prefetch_b8
      v16bf af = ldA(xr + kt * 32 + abase);
      v16bf bf = ldB(wr + kt * 32 + abase);
      acc = wmma_bf16(af, bf, acc);
    }
#pragma unroll
    for (int r = 0; r < 8; ++r) {
      int nl = r + hi16 * 8;
      int p  = nt * 16 + nl;
      if (p < NPIX) {
        float y = (acc[r] + b_p[o]) * s_p[o] + t_p[o];
        out[((size_t)b * DIMC + o) * NPIX + p] = y;
      }
    }
  }
}

// ======================================================================
extern "C" void kernel_launch(void* const* d_in, const int* in_sizes, int n_in,
                              void* d_out, int out_size, void* d_ws, size_t ws_size,
                              hipStream_t stream) {
  const float* hidden = (const float*)d_in[0];
  const float* w_q = (const float*)d_in[1];
  const float* b_q = (const float*)d_in[2];
  const float* s_q = (const float*)d_in[3];
  const float* t_q = (const float*)d_in[4];
  const float* w_k = (const float*)d_in[5];
  const float* b_k = (const float*)d_in[6];
  const float* s_k = (const float*)d_in[7];
  const float* t_k = (const float*)d_in[8];
  const float* w_v = (const float*)d_in[9];
  const float* b_v = (const float*)d_in[10];
  const float* s_v = (const float*)d_in[11];
  const float* t_v = (const float*)d_in[12];
  const float* w_vl = (const float*)d_in[13];
  const float* b_vl = (const float*)d_in[14];
  const float* s_vl = (const float*)d_in[15];
  const float* t_vl = (const float*)d_in[16];
  const float* w_th1 = (const float*)d_in[17];
  const float* b_th1 = (const float*)d_in[18];
  const float* w_th2 = (const float*)d_in[19];
  const float* b_th2 = (const float*)d_in[20];
  const float* w_p = (const float*)d_in[21];
  const float* b_p = (const float*)d_in[22];
  const float* s_p = (const float*)d_in[23];
  const float* t_p = (const float*)d_in[24];
  const float* attn_biases = (const float*)d_in[25];
  const int*   bias_idxs   = (const int*)d_in[26];
  float* out = (float*)d_out;

  // workspace layout
  char* ws = (char*)d_ws;
  const size_t QB_SZ   = (size_t)BB * HH * N2 * KDIM * sizeof(__bf16);   // 13.6 MB
  const size_t VIMG_SZ = (size_t)BB * DHC * NPIX * sizeof(float);        // 102.8 MB
  const size_t VB_SZ   = (size_t)BB * DHC * NP * sizeof(__bf16);         // 58.7 MB
  const size_t XB_SZ   = (size_t)BB * N2 * DIMC * sizeof(__bf16);        // 13.6 MB
  const size_t WQKV_SZ = (size_t)1536 * DIMC * sizeof(__bf16);           // 0.79 MB
  size_t off = 0;
  __bf16* qb   = (__bf16*)(ws + off); off += QB_SZ;
  __bf16* kb   = (__bf16*)(ws + off); off += QB_SZ;
  float*  vimg = (float*)(ws + off);  off += VIMG_SZ;
  __bf16* vb   = (__bf16*)(ws + off); off += VB_SZ;
  __bf16* xb   = (__bf16*)(ws + off); off += XB_SZ;
  __bf16* wqkv = (__bf16*)(ws + off); off += WQKV_SZ;
  __bf16* wpb  = (__bf16*)(ws + off);

  const int noff = in_sizes[25] / HH;   // 196 unique offsets

  prep_wp_kernel<<<(DIMC * DHC) / 256, 256, 0, stream>>>(w_p, wpb);
  prep_wqkv_kernel<<<(1536 * DIMC) / 256, 256, 0, stream>>>(w_q, w_k, w_v, wqkv);
  prep_x_kernel<<<(BB * N2 * DIMC) / 256, 256, 0, stream>>>(hidden, xb);
  prep_vpad_kernel<<<(BB * DHC * (NP - NPIX)) / 256, 256, 0, stream>>>(vb);

  qkv_kernel<<<dim3(13, 12, BB), 256, 0, stream>>>(
      xb, wqkv, b_q, s_q, t_q, b_k, s_k, t_k, b_v, s_v, t_v,
      qb, kb, vimg, vb);

  const size_t smem = 114688u * 2u + 57344u;   // attnA + attnB + attnC = 280 KB
  attn_kernel<<<dim3(13, BB), 256, smem, stream>>>(
      qb, kb, vimg, vb, wpb, bias_idxs, attn_biases, noff,
      w_th1, b_th1, w_th2, b_th2, w_vl, b_vl, s_vl, t_vl,
      b_p, s_p, t_p, out);
}
